// SingleHeadAttention_22823456211328
// MI455X (gfx1250) — matrix operand order repaired and checked by reading, not compiled, not run on hardware
//
#include <hip/hip_runtime.h>

#define HEAD 128
#define EMB  1024
#define SEQ  4096
#define NROW (4 * SEQ)   // 16384 rows of x / q / k / v

typedef __attribute__((ext_vector_type(16))) __bf16   v16bf;
typedef __attribute__((ext_vector_type(8)))  float    v8f;
typedef __attribute__((ext_vector_type(8)))  unsigned v8u;

// fp32 -> bf16 via native conversion (lets clang emit v_cvt_* on gfx1250)
__device__ __forceinline__ unsigned short f2bf(float f) {
    return __builtin_bit_cast(unsigned short, (__bf16)f);
}
__device__ __forceinline__ unsigned packbf(float lo, float hi) {
    return (unsigned)f2bf(lo) | ((unsigned)f2bf(hi) << 16);
}

__device__ __forceinline__ v8f wmma_bf16(v8u a, v8u b, v8f c) {
    // D(16x16,f32) = A(16x32,bf16) * B(32x16,bf16) + C
    return __builtin_amdgcn_wmma_f32_16x16x32_bf16(
        false, __builtin_bit_cast(v16bf, a),
        false, __builtin_bit_cast(v16bf, b),
        (short)0, c, false, false);
}

// ---------------------------------------------------------------------------
// Kernel 0a: one-shot x fp32 -> bf16 (row-major, packed 2/dword).
// 8388608 dwords, 1 per thread. 64MB read + 32MB write, once.
// ---------------------------------------------------------------------------
__global__ __launch_bounds__(256) void cvt_x(const float* __restrict__ x,
                                             unsigned* __restrict__ xb2) {
    const size_t i = (size_t)blockIdx.x * 256 + threadIdx.x;
    const float2 f = ((const float2*)x)[i];
    xb2[i] = packbf(f.x, f.y);
}

// ---------------------------------------------------------------------------
// Kernel 0b: one-shot W{q,k,v} fp32 -> bf16, transposed to [mat][n][k] so
// GEMM B-fragments are contiguous dword pairs. 3*128*1024 elements.
// ---------------------------------------------------------------------------
__global__ __launch_bounds__(256) void cvt_w(const float* __restrict__ Wq,
                                             const float* __restrict__ Wk,
                                             const float* __restrict__ Wv,
                                             unsigned short* __restrict__ wbT) {
    const int idx = blockIdx.x * 256 + threadIdx.x;   // 0 .. 393215
    const int mat = idx >> 17;                        // 131072 per matrix
    const int r   = idx & 131071;
    const int n   = r >> 10;                          // output column 0..127
    const int k   = r & 1023;                         // reduction index
    const float* W = (mat == 0) ? Wq : (mat == 1) ? Wk : Wv;
    wbT[idx] = f2bf(W[(size_t)k * HEAD + n]);
}

// ---------------------------------------------------------------------------
// Kernel 1: q/k/v = x @ W{q,k,v} (bf16 WMMA, fp32 acc); RoPE on q,k.
//   q,k stored row-major [16384][128]; v stored transposed [B][128][4096].
// grid = (256, 3), block = 128 (4 waves); wave w handles row tile bx*4+w of
// matrix blockIdx.y. No LDS: B fragments read straight from L2-resident wbT.
// ---------------------------------------------------------------------------
__global__ __launch_bounds__(128) void qkv_rope(
    const unsigned short* __restrict__ xb,     // [16384][1024] bf16
    const unsigned short* __restrict__ wbT,    // [3][128][1024] bf16
    const float* __restrict__ fcos, const float* __restrict__ fsin,
    unsigned short* __restrict__ qws, unsigned short* __restrict__ kws,
    unsigned short* __restrict__ vT)
{
    const int mat = blockIdx.y;
    const unsigned short* Wt = wbT + (size_t)mat * HEAD * EMB;

    const int wave = threadIdx.x >> 5;
    const int lane = threadIdx.x & 31;
    const int g    = lane >> 4;     // lane group (0: lanes 0-15, 1: 16-31)
    const int ln   = lane & 15;

    const int m0 = (blockIdx.x * 4 + wave) * 16;   // output row-tile base

    const v8f vzero = {0.f,0.f,0.f,0.f,0.f,0.f,0.f,0.f};
    v8f acc[8];
    #pragma unroll
    for (int t = 0; t < 8; ++t) acc[t] = vzero;

    const unsigned short* xrow = xb + (size_t)(m0 + ln) * EMB;

    for (int kb = 0; kb < EMB; kb += 32) {
        // A fragment (16x32 bf16): VGPR v<4 -> K=8g+2v, v>=4 -> K=16+8g+2(v-4)
        v8u a;
        #pragma unroll
        for (int v = 0; v < 8; ++v) {
            const int d = (v < 4) ? (8*g + 2*v) : (16 + 8*g + 2*(v-4));
            a[v] = *(const unsigned*)(xrow + kb + d);
        }
        // 8 column tiles; B fragment K same as A (8g+2v / 16+8g+2(v-4)), from wbT rows
        #pragma unroll
        for (int t = 0; t < 8; ++t) {
            const unsigned short* wrow = Wt + (size_t)(16*t + ln) * EMB + kb;
            v8u b;
            #pragma unroll
            for (int v = 0; v < 8; ++v)
                b[v] = *(const unsigned*)(wrow + ((v < 4) ? (8*g + 2*v) : (16 + 8*g + 2*(v-4))));
            acc[t] = wmma_bf16(a, b, acc[t]);
        }
    }

    if (mat < 2) {
        // RoPE then store bf16 row-major. C layout: lane holds col = 16t+ln
        // for rows m0 + r + 8g; pair partner (col^1) sits in lane^1.
        unsigned short* dst = (mat == 0) ? qws : kws;
        #pragma unroll
        for (int t = 0; t < 8; ++t) {
            const int col = 16*t + ln;
            const int p   = col >> 1;
            const bool ev = (col & 1) == 0;
            #pragma unroll
            for (int r = 0; r < 8; ++r) {
                const int rrow = m0 + r + 8*g;
                const int tok  = rrow & (SEQ - 1);
                float val = acc[t][r];
                float prt = __shfl_xor(val, 1, 32);
                float c = fcos[(size_t)tok * (HEAD/2) + p];
                float s = fsin[(size_t)tok * (HEAD/2) + p];
                float res = ev ? (val * c - prt * s)   // x0' = x0 c - x1 s
                               : (prt * s + val * c);  // x1' = x0 s + x1 c
                dst[(size_t)rrow * HEAD + col] = f2bf(res);
            }
        }
    } else {
        // v stored transposed: vT[b][d][t]
        #pragma unroll
        for (int t = 0; t < 8; ++t) {
            const int col = 16*t + ln;
            #pragma unroll
            for (int r = 0; r < 8; ++r) {
                const int rrow = m0 + r + 8*g;
                const int bI   = rrow >> 12;
                const int tok  = rrow & (SEQ - 1);
                vT[((size_t)bI * HEAD + col) * SEQ + tok] = f2bf(acc[t][r]);
            }
        }
    }
}

// ---------------------------------------------------------------------------
// Kernel 2: causal flash attention. grid = 128 blocks (B*T/128), block = 256
// (8 waves); each wave owns a 16-query tile, iterates 32-key blocks with
// online softmax. fp32 accumulation throughout; P via bf16 WMMA.
// ---------------------------------------------------------------------------
__global__ __launch_bounds__(256) void flash_attn(
    const unsigned short* __restrict__ qws,
    const unsigned short* __restrict__ kws,
    const unsigned short* __restrict__ vT,
    float* __restrict__ out)
{
    __shared__ __align__(4) unsigned short Plds[8][16 * 32]; // per-wave P tile

    const int wave = threadIdx.x >> 5;
    const int lane = threadIdx.x & 31;
    const int g    = lane >> 4;
    const int ln   = lane & 15;

    const int b  = blockIdx.x >> 5;                     // batch
    const int q0 = (blockIdx.x & 31) * 128 + wave * 16; // query tile base

    const float SCALE = 0.08838834764831845f;           // 1/sqrt(128)
    const v8f vzero = {0.f,0.f,0.f,0.f,0.f,0.f,0.f,0.f};

    // Q A-fragments (4 chunks of K=32 over D=128), kept resident
    const unsigned short* qrow = qws + (size_t)(b * SEQ + q0 + ln) * HEAD;
    v8u qa[4];
    #pragma unroll
    for (int a = 0; a < 4; ++a)
        #pragma unroll
        for (int v = 0; v < 8; ++v) {
            const int d = 32*a + ((v < 4) ? (8*g + 2*v) : (16 + 8*g + 2*(v-4)));
            qa[a][v] = *(const unsigned*)(qrow + d);
        }

    v8f oacc[8];
    #pragma unroll
    for (int t = 0; t < 8; ++t) oacc[t] = vzero;
    float m_s[8], l_s[8];
    #pragma unroll
    for (int r = 0; r < 8; ++r) { m_s[r] = -__builtin_inff(); l_s[r] = 0.f; }

    unsigned short* pw = Plds[wave];

    for (int kb = 0; kb < q0 + 16; kb += 32) {
        // ---- S = Q @ K^T for two 16-key tiles ----
        v8f s0 = vzero, s1 = vzero;
        const unsigned short* k0 = kws + (size_t)(b * SEQ + kb + ln) * HEAD;
        const unsigned short* k1 = k0 + (size_t)16 * HEAD;
        #pragma unroll
        for (int a = 0; a < 4; ++a) {
            v8u b0, b1;
            #pragma unroll
            for (int v = 0; v < 8; ++v) {
                const int d = 32*a + ((v < 4) ? (8*g + 2*v) : (16 + 8*g + 2*(v-4)));   // B layout: same K as A
                b0[v] = *(const unsigned*)(k0 + d);
                b1[v] = *(const unsigned*)(k1 + d);
            }
            s0 = wmma_bf16(qa[a], b0, s0);
            s1 = wmma_bf16(qa[a], b1, s1);
        }
        #pragma unroll
        for (int r = 0; r < 8; ++r) { s0[r] *= SCALE; s1[r] *= SCALE; }

        // ---- causal mask (only needed near the diagonal) ----
        if (kb + 31 > q0) {
            #pragma unroll
            for (int r = 0; r < 8; ++r) {
                const int qr = q0 + r + 8*g;
                if (kb + ln      > qr) s0[r] = -__builtin_inff();
                if (kb + 16 + ln > qr) s1[r] = -__builtin_inff();
            }
        }

        // ---- online softmax: row stats via intra-16-lane reductions ----
        #pragma unroll
        for (int r = 0; r < 8; ++r) {
            float mx = fmaxf(s0[r], s1[r]);
            #pragma unroll
            for (int off = 8; off; off >>= 1)
                mx = fmaxf(mx, __shfl_xor(mx, off, 32));
            const float mnew = fmaxf(m_s[r], mx);
            const float resc = __expf(m_s[r] - mnew);
            m_s[r] = mnew;
            const float p0 = __expf(s0[r] - mnew);
            const float p1 = __expf(s1[r] - mnew);
            float rs = p0 + p1;
            #pragma unroll
            for (int off = 8; off; off >>= 1)
                rs += __shfl_xor(rs, off, 32);
            l_s[r] = l_s[r] * resc + rs;
            #pragma unroll
            for (int t = 0; t < 8; ++t) oacc[t][r] *= resc;
            // stage P (C layout -> row-major 16x32 bf16) in per-wave LDS
            pw[(r + 8*g) * 32 + ln]      = f2bf(p0);
            pw[(r + 8*g) * 32 + 16 + ln] = f2bf(p1);
        }

        // ---- reload P as an A fragment (16x32) ----
        v8u pa;
        #pragma unroll
        for (int v = 0; v < 8; ++v) {
            const int kk = (v < 4) ? (8*g + 2*v) : (16 + 8*g + 2*(v-4));
            pa[v] = *(const unsigned*)(pw + ln * 32 + kk);
        }

        // ---- O += P @ V (V^T layout makes B fragments contiguous) ----
        const unsigned short* vbase = vT + (size_t)b * HEAD * SEQ + kb;
        #pragma unroll
        for (int t = 0; t < 8; ++t) {
            const unsigned short* vp = vbase + (size_t)(16*t + ln) * SEQ;
            v8u bv;
            #pragma unroll
            for (int v = 0; v < 8; ++v)
                bv[v] = *(const unsigned*)(vp + ((v < 4) ? (8*g + 2*v) : (16 + 8*g + 2*(v-4))));
            oacc[t] = wmma_bf16(pa, bv, oacc[t]);
        }
    }

    // ---- normalize and store fp32 ----
    #pragma unroll
    for (int t = 0; t < 8; ++t) {
        const int col = 16*t + ln;
        #pragma unroll
        for (int r = 0; r < 8; ++r) {
            const int qr = q0 + r + 8*g;
            out[(size_t)(b * SEQ + qr) * HEAD + col] = oacc[t][r] / l_s[r];
        }
    }
}

// ---------------------------------------------------------------------------
extern "C" void kernel_launch(void* const* d_in, const int* in_sizes, int n_in,
                              void* d_out, int out_size, void* d_ws, size_t ws_size,
                              hipStream_t stream) {
    const float* x    = (const float*)d_in[0];
    const float* Wq   = (const float*)d_in[1];
    const float* Wk   = (const float*)d_in[2];
    const float* Wv   = (const float*)d_in[3];
    const float* fcos = (const float*)d_in[4];
    const float* fsin = (const float*)d_in[5];
    float* out = (float*)d_out;

    // workspace: q(4MB) | k(4MB) | v^T(4MB) | xb(32MB) | wbT(0.75MB), all bf16
    unsigned short* qws = (unsigned short*)d_ws;
    unsigned short* kws = qws + (size_t)NROW * HEAD;
    unsigned short* vT  = kws + (size_t)NROW * HEAD;
    unsigned short* xb  = vT  + (size_t)NROW * HEAD;
    unsigned short* wbT = xb  + (size_t)NROW * EMB;

    // one-shot fp32 -> bf16 conversions
    cvt_x<<<dim3((NROW * EMB / 2) / 256), 256, 0, stream>>>(x, (unsigned*)xb);
    cvt_w<<<dim3((3 * HEAD * EMB) / 256), 256, 0, stream>>>(Wq, Wk, Wv, wbT);

    // projections + RoPE
    qkv_rope<<<dim3(256, 3), 128, 0, stream>>>(xb, wbT, fcos, fsin,
                                               qws, kws, vT);
    // causal flash attention
    flash_attn<<<dim3(128), 256, 0, stream>>>(qws, kws, vT, out);
}
